// TGP_SSID_SOTA_10883447128187
// MI455X (gfx1250) — matrix operand order repaired and checked by reading, not compiled, not run on hardware
//
#include <hip/hip_runtime.h>
#include <hip/hip_bf16.h>

// ---------------- CDNA5 WMMA helpers (wave32, 16x16x32 bf16) ----------------
typedef __attribute__((ext_vector_type(16))) __bf16 v16bf;
typedef __attribute__((ext_vector_type(8)))  float  v8f;

__device__ __forceinline__ v16bf load_frag2(const __bf16* p0, const __bf16* p1) {
    v16bf r;
    ((uint4*)&r)[0] = *(const uint4*)p0;   // elements 0..7
    ((uint4*)&r)[1] = *(const uint4*)p1;   // elements 8..15
    return r;
}

__device__ __forceinline__ v8f wmma_bf16(v16bf a, v16bf b, v8f c) {
    return __builtin_amdgcn_wmma_f32_16x16x32_bf16(false, a, false, b, (short)0, c, false, false);
}

// A fragment (16x32 bf16, row-major source, row stride = KDIM):
//   lane: m = lane&15 (row), g = lane>>4; elems0..15 = K kb+16g .. +16 (contiguous)
__device__ __forceinline__ v16bf load_afrag(const __bf16* arow, int kb, int g) {
    return load_frag2(arow + kb + 16 * g, arow + kb + 16 * g + 8);
}
// B fragment (32x16 bf16) from column-major B (BT[n][k] row-major, row stride KDIM):
//   lane: n = lane&15 (col), g = lane>>4; elems0..15 = K kb+16g .. +16 (contiguous)
__device__ __forceinline__ v16bf load_bfrag(const __bf16* bcol, int kb, int g) {
    const __bf16* p = bcol + kb + 16 * g;
    return load_frag2(p, p + 8);
}

// -------- gfx1250 async global->LDS copy (ASYNCcnt path), inline asm --------
// LDS dest operand = wave-relative LDS byte address = low 32 bits of the
// generic pointer to a __shared__ object (ISA: LDS aperture).
__device__ __forceinline__ void async_copy_b128(unsigned ldsDst, const void* gsrc) {
    asm volatile("global_load_async_to_lds_b128 %0, %1, off"
                 :: "v"(ldsDst), "v"(gsrc)
                 : "memory");
}
__device__ __forceinline__ void wait_async0() {
    asm volatile("s_wait_asynccnt 0x0" ::: "memory");
}
__device__ __forceinline__ unsigned lds_addr32(const void* p) {
    return (unsigned)(unsigned long long)p;
}

// ---------------- prep: transpose + bf16-convert weights, proto norms -------
__global__ void prep_weights_kernel(const float* __restrict__ w1, const float* __restrict__ w2,
                                    const float* __restrict__ w3, const float* __restrict__ protos,
                                    __bf16* __restrict__ w1T, __bf16* __restrict__ w2T,
                                    __bf16* __restrict__ w3T, __bf16* __restrict__ pB,
                                    float* __restrict__ pn) {
    int i = blockIdx.x * blockDim.x + threadIdx.x;          // grid covers 131072
    { int n = i >> 8, k = i & 255; w1T[i] = (__bf16)w1[k * 512 + n]; }      // [512][256]
    { int n = i >> 9, k = i & 511; w2T[i] = (__bf16)w2[k * 256 + n]; }      // [256][512]
    if (i < 32768) { int n = i >> 8, k = i & 255; w3T[i] = (__bf16)w3[k * 128 + n]; } // [128][256]
    if (i < 32768) {
        int p = i >> 7, k = i & 127;
        pB[i] = (p < 250) ? (__bf16)protos[p * 128 + k] : (__bf16)0.0f;     // [256][128] padded
    }
    if (i < 256) {
        float s = 0.0f;
        if (i < 250) {
            for (int k = 0; k < 128; ++k) { float v = protos[i * 128 + k]; s += v * v; }
        }
        pn[i] = s;
    }
}

__global__ void convert_x_kernel(const float* __restrict__ x, __bf16* __restrict__ xb, int n4) {
    int i = blockIdx.x * blockDim.x + threadIdx.x;
    if (i < n4) {
        float4 v = ((const float4*)x)[i];
        __bf16 t[4] = {(__bf16)v.x, (__bf16)v.y, (__bf16)v.z, (__bf16)v.w};
        ((uint2*)xb)[i] = *(uint2*)t;
    }
}

// ---------------- fused GEMM + bias + LayerNorm + ReLU ----------------------
// Block = 256 threads = 8 waves; each wave owns 16 rows x NDIM cols, accum in VGPRs.
// B (weights) double-buffered in LDS via async global->LDS copies: one 32-K
// slice [NDIM][32] bf16 per buffer, shared by all 8 waves (8x less L2 traffic).
template <int KDIM, int NDIM>
__global__ __launch_bounds__(256) void gemm_ln_relu_kernel(
    const __bf16* __restrict__ A, const __bf16* __restrict__ BT,
    const float* __restrict__ bias, const float* __restrict__ gam,
    const float* __restrict__ bet, __bf16* __restrict__ Out) {
    constexpr int KT = KDIM / 32;
    constexpr int NT = NDIM / 16;
    __shared__ __align__(16) __bf16 stageB[2][NDIM][32];   // 64KB (NDIM=512) / 32KB

    const int tid = threadIdx.x;
    const int lane = tid & 31, wave = tid >> 5;
    const int rowBase = blockIdx.x * 128 + wave * 16;
    const int m = lane & 15, g = lane >> 4;

    // async-stage one 32-wide K slice of BT into LDS buffer `buf`
    auto issue_copies = [&](int buf, int kt) {
        constexpr int CPT = (NDIM * 4) / 256;              // 16B chunks per thread
#pragma unroll
        for (int i = 0; i < CPT; ++i) {
            int chunk = i * 256 + tid;                     // 0 .. NDIM*4-1
            int n = chunk >> 2, part = chunk & 3;
            const __bf16* src = BT + (size_t)n * KDIM + kt * 32 + part * 8;
            async_copy_b128(lds_addr32(&stageB[buf][n][part * 8]), src);
        }
    };

    issue_copies(0, 0);

    const __bf16* arow = A + (size_t)(rowBase + m) * KDIM;
    v8f acc[NT];
    const v8f vzero = {0.f, 0.f, 0.f, 0.f, 0.f, 0.f, 0.f, 0.f};
#pragma unroll
    for (int nt = 0; nt < NT; ++nt) acc[nt] = vzero;

    wait_async0();
    __syncthreads();

    for (int kt = 0; kt < KT; ++kt) {
        const int buf = kt & 1;
        if (kt + 1 < KT) issue_copies(buf ^ 1, kt + 1);    // prefetch next slice
        v16bf a = load_afrag(arow, kt * 32, g);
#pragma unroll
        for (int nt = 0; nt < NT; ++nt) {
            const __bf16* bp = &stageB[buf][nt * 16 + m][16 * g];
            v16bf b = load_frag2(bp, bp + 8);              // ds_load_b128 x2
            acc[nt] = wmma_bf16(a, b, acc[nt]);
        }
        if (kt + 1 < KT) {
            wait_async0();
            __syncthreads();
        }
    }

    // bias + LayerNorm stats. Lane holds rows 8g+r (r=0..7), cols nt*16+m.
    float sum[8], sq[8];
#pragma unroll
    for (int r = 0; r < 8; ++r) { sum[r] = 0.f; sq[r] = 0.f; }
#pragma unroll
    for (int nt = 0; nt < NT; ++nt) {
        float bv = bias[nt * 16 + m];
#pragma unroll
        for (int r = 0; r < 8; ++r) {
            float v = acc[nt][r] + bv;
            acc[nt][r] = v;
            sum[r] += v;
            sq[r] += v * v;
        }
    }
    float mu[8], rs[8];
#pragma unroll
    for (int r = 0; r < 8; ++r) {
#pragma unroll
        for (int s = 1; s < 16; s <<= 1) {
            sum[r] += __shfl_xor(sum[r], s, 32);
            sq[r]  += __shfl_xor(sq[r],  s, 32);
        }
        float mm = sum[r] * (1.0f / NDIM);
        float var = sq[r] * (1.0f / NDIM) - mm * mm;
        mu[r] = mm;
        rs[r] = rsqrtf(var + 1e-5f);
    }
    // normalize + affine + ReLU + store bf16
#pragma unroll
    for (int nt = 0; nt < NT; ++nt) {
        int col = nt * 16 + m;
        float gv = gam[col], bev = bet[col];
#pragma unroll
        for (int r = 0; r < 8; ++r) {
            float v = (acc[nt][r] - mu[r]) * rs[r] * gv + bev;
            v = fmaxf(v, 0.0f);
            Out[(size_t)(rowBase + 8 * g + r) * NDIM + col] = (__bf16)v;
        }
    }
}

// ---------------- head: GEMM3 + ReLU + cdist(protos) + top-15 ---------------
// Block = 128 threads = 4 waves; each wave owns 16 rows. 64KB static LDS.
__global__ __launch_bounds__(128) void head_kernel(
    const __bf16* __restrict__ H2, const __bf16* __restrict__ W3T,
    const float* __restrict__ b3, const __bf16* __restrict__ PB,
    const float* __restrict__ pn, float* __restrict__ out) {
    __shared__ __align__(16) float dbuf[4][16][256];   // per-wave 16KB; z staged in first 4KB

    const int lane = threadIdx.x & 31, wave = threadIdx.x >> 5;
    const int rowBase = blockIdx.x * 64 + wave * 16;
    const int m = lane & 15, g = lane >> 4;
    const v8f vzero = {0.f, 0.f, 0.f, 0.f, 0.f, 0.f, 0.f, 0.f};

    // ---- stage 1: z = relu(H2 @ w3 + b3), K=256 (8 k-steps), 128 cols (8 n-tiles)
    v8f acc[8];
#pragma unroll
    for (int nt = 0; nt < 8; ++nt) acc[nt] = vzero;
    const __bf16* arow = H2 + (size_t)(rowBase + m) * 256;
#pragma unroll
    for (int kt = 0; kt < 8; ++kt) {
        v16bf a = load_afrag(arow, kt * 32, g);
#pragma unroll
        for (int nt = 0; nt < 8; ++nt) {
            v16bf b = load_bfrag(W3T + (size_t)(nt * 16 + m) * 256, kt * 32, g);
            acc[nt] = wmma_bf16(a, b, acc[nt]);
        }
    }

    // bias + relu; accumulate ||z||^2 partials; stage z (bf16) to LDS in A-layout order
    __bf16* zb = (__bf16*)&dbuf[wave][0][0];           // [16][128] bf16
    float znp[8];
#pragma unroll
    for (int r = 0; r < 8; ++r) znp[r] = 0.f;
#pragma unroll
    for (int nt = 0; nt < 8; ++nt) {
        int col = nt * 16 + m;
        float bv = b3[col];
#pragma unroll
        for (int r = 0; r < 8; ++r) {
            float z = fmaxf(acc[nt][r] + bv, 0.0f);
            znp[r] += z * z;
            zb[(8 * g + r) * 128 + col] = (__bf16)z;
        }
    }
#pragma unroll
    for (int r = 0; r < 8; ++r) {
#pragma unroll
        for (int s = 1; s < 16; s <<= 1) znp[r] += __shfl_xor(znp[r], s, 32);
    }

    // ---- stage 2: s = z @ protos^T, K=128 (4 k-steps), 256 cols (16 n-tiles, 250 valid)
    v8f sm[16];
#pragma unroll
    for (int nt = 0; nt < 16; ++nt) sm[nt] = vzero;
    const __bf16* zrow = zb + m * 128;
#pragma unroll
    for (int kt = 0; kt < 4; ++kt) {
        v16bf a = load_afrag(zrow, kt * 32, g);        // LDS reads, wave-local in-order
#pragma unroll
        for (int nt = 0; nt < 16; ++nt) {
            v16bf b = load_bfrag(PB + (size_t)(nt * 16 + m) * 128, kt * 32, g);
            sm[nt] = wmma_bf16(a, b, sm[nt]);
        }
    }

    // ---- distances -> LDS (overwrites z staging; all z reads already issued)
#pragma unroll
    for (int nt = 0; nt < 16; ++nt) {
        int col = nt * 16 + m;
        float pc = pn[col];
#pragma unroll
        for (int r = 0; r < 8; ++r) {
            float d2 = znp[r] + pc - 2.0f * sm[nt][r];
            float d = sqrtf(fmaxf(d2, 0.0f));
            if (col >= 250) d = 3.0e38f;
            dbuf[wave][8 * g + r][col] = d;
        }
    }

    // ---- top-15 of 250: 2 lanes per row, each scans 128, branchless sorted insert
    float t[15];
#pragma unroll
    for (int i = 0; i < 15; ++i) t[i] = 3.0e38f;
    const float* dr = &dbuf[wave][m][g * 128];
    for (int j = 0; j < 128; ++j) {
        float v = dr[j];
        if (v < t[14]) {
            float c = v;
#pragma unroll
            for (int i = 0; i < 15; ++i) {
                float a = t[i];
                t[i] = fminf(a, c);
                c = fmaxf(a, c);
            }
        }
    }
    // merge the two sorted half-lists: 15 smallest of union = min(t[i], partner[14-i])
    float ssum = 0.0f;
#pragma unroll
    for (int i = 0; i < 15; ++i) ssum += fminf(t[i], __shfl_xor(t[14 - i], 16, 32));
    float mn = fminf(t[0], __shfl_xor(t[0], 16, 32));

    float score = 0.1f * (ssum * (1.0f / 15.0f)) + 0.9f * mn;
    if (lane < 16) out[rowBase + m] = 1.0f - expf(-0.3f * score);
}

// ---------------- launch ----------------------------------------------------
extern "C" void kernel_launch(void* const* d_in, const int* in_sizes, int n_in,
                              void* d_out, int out_size, void* d_ws, size_t ws_size,
                              hipStream_t stream) {
    const float* x      = (const float*)d_in[0];
    const float* w1     = (const float*)d_in[1];
    const float* b1     = (const float*)d_in[2];
    const float* g1     = (const float*)d_in[3];
    const float* be1    = (const float*)d_in[4];
    const float* w2     = (const float*)d_in[5];
    const float* b2     = (const float*)d_in[6];
    const float* g2     = (const float*)d_in[7];
    const float* be2    = (const float*)d_in[8];
    const float* w3     = (const float*)d_in[9];
    const float* b3     = (const float*)d_in[10];
    const float* protos = (const float*)d_in[11];
    float* out = (float*)d_out;

    char* ws = (char*)d_ws;
    __bf16* w1T = (__bf16*)(ws + 0);                    // 512*256*2 = 256KB
    __bf16* w2T = (__bf16*)(ws + 262144);               // 256*512*2 = 256KB
    __bf16* w3T = (__bf16*)(ws + 524288);               // 128*256*2 = 64KB
    __bf16* pB  = (__bf16*)(ws + 589824);               // 256*128*2 = 64KB (padded)
    float*  pn  = (float*)(ws + 655360);                // 256*4
    __bf16* xh2 = (__bf16*)(ws + (1u << 20));           // [N,256] bf16: x, later h2 (32MB)
    __bf16* h1  = (__bf16*)(ws + (1u << 20) + (size_t)33554432);  // [N,512] bf16 (64MB)

    const int N = 65536;

    prep_weights_kernel<<<512, 256, 0, stream>>>(w1, w2, w3, protos, w1T, w2T, w3T, pB, pn);
    convert_x_kernel<<<(N * 256 / 4 + 255) / 256, 256, 0, stream>>>(x, xh2, N * 256 / 4);
    gemm_ln_relu_kernel<256, 512><<<N / 128, 256, 0, stream>>>(xh2, w1T, b1, g1, be1, h1);
    gemm_ln_relu_kernel<512, 256><<<N / 128, 256, 0, stream>>>(h1, w2T, b2, g2, be2, xh2);
    head_kernel<<<N / 64, 128, 0, stream>>>(xh2, w3T, b3, pB, pn, out);
}
